// Kernel_7886596238811945167_41377714930046
// MI455X (gfx1250) — matrix operand order repaired and checked by reading, not compiled, not run on hardware
//
#include <hip/hip_runtime.h>
#include <hip/hip_bf16.h>
#include <cstdint>
#include <cstddef>

typedef __attribute__((ext_vector_type(16))) _Float16 v16h;
typedef __attribute__((ext_vector_type(8)))  _Float16 v8h;
typedef __attribute__((ext_vector_type(8)))  float    v8f;

#define C_DIM   256
#define HW_DIM  3136
#define NBATCH  32
#define PI_HALF 1.57079632679489662f
#define RS_HW   (1.0f/56.0f)   // 1/sqrt(3136)
#define RS_C    (1.0f/16.0f)   // 1/sqrt(256)

__device__ __forceinline__ v16h mk16(v8h lo, v8h hi) {
  v16h r;
#pragma unroll
  for (int i = 0; i < 8; ++i) { r[i] = lo[i]; r[i + 8] = hi[i]; }
  return r;
}

// ---- CDNA5 async global->LDS copy (16 B per lane), tracked by ASYNCcnt ----
// LDS byte address = low 32 bits of the generic shared pointer.
__device__ __forceinline__ void async_copy16(const void* gsrc, void* ldst) {
  unsigned l = (unsigned)(uintptr_t)ldst;
  asm volatile("global_load_async_to_lds_b128 %0, %1, off"
               :: "v"(l), "v"(gsrc) : "memory");
}
__device__ __forceinline__ void wait_async0() {
  asm volatile("s_wait_asynccnt 0" ::: "memory");
}

// ---------------- small f32 -> f16 conversion (Wp3) ----------------
__global__ void cvt_f32_f16(const float* __restrict__ src, _Float16* __restrict__ dst, int n) {
  int i = blockIdx.x * blockDim.x + threadIdx.x;
  if (i < n) dst[i] = (_Float16)src[i];
}

// ---------------- grouped 1x1 convs (G=8, CG=32) -> lhs/rhs f16 ----------------
__global__ __launch_bounds__(64)
void prep_kernel(const float* __restrict__ x,
                 const float* __restrict__ Wg1, const float* __restrict__ bg1,
                 const float* __restrict__ Wg2, const float* __restrict__ bg2,
                 _Float16* __restrict__ lhs, _Float16* __restrict__ rhs)
{
  int hw = blockIdx.x * 64 + threadIdx.x;  // 49*64 == 3136
  int c  = blockIdx.y;
  int n  = blockIdx.z;
  int g = c >> 5, o = c & 31;
  const float* xb = x + ((size_t)n * C_DIM + (size_t)g * 32) * HW_DIM + hw;
  const float* w1 = Wg1 + g * 1024 + o * 32;
  const float* w2 = Wg2 + g * 1024 + o * 32;
  float a1 = 0.f, a2 = 0.f;
#pragma unroll
  for (int i = 0; i < 32; ++i) {
    float xv = xb[(size_t)i * HW_DIM];
    a1 = fmaf(w1[i], xv, a1);
    a2 = fmaf(w2[i], xv, a2);
  }
  size_t oidx = ((size_t)n * C_DIM + c) * HW_DIM + hw;
  lhs[oidx] = (_Float16)(a1 + bg1[c]);
  rhs[oidx] = (_Float16)(a2 + bg2[c]);
}

// ---------------- WMMA GEMM, 128x64 block tile, BLOCK_K=32, double-buffered ----------------
// MODE 0: t6 = max(Wp3@x + bp3, sin(x*pi/2))      A=Wp3h[256,256] B=x(f32,K-major) -> t6h f16
// MODE 1: t7t = (p7w * (lhs@rhs^T)/sqrt(hw))^T    A=lhs[256,3136] B=rhs(N-major)   -> t7t f16
// MODE 2: out = (t7^T @ t6)/sqrt(C)               A=t7t[256,256]  B=t6h(K-major)   -> out f32
template<int MODE>
__global__ __launch_bounds__(256)
void gemm_wmma(const _Float16* __restrict__ Aall,
               const _Float16* __restrict__ Bhall,
               const float*    __restrict__ Bfall,
               const float*    __restrict__ xf,
               const float*    __restrict__ bp3,
               const float*    __restrict__ p7w,
               _Float16*       __restrict__ outH,
               float*          __restrict__ outF)
{
  const int K   = (MODE == 1) ? HW_DIM : C_DIM;
  const int lda = (MODE == 1) ? HW_DIM : C_DIM;
  const int ldb = HW_DIM;
  const size_t aStride = (MODE == 0) ? 0
                        : (MODE == 1) ? (size_t)C_DIM * HW_DIM
                                      : (size_t)C_DIM * C_DIM;
  const size_t bStride = (size_t)C_DIM * HW_DIM;

  const int batch = blockIdx.z;
  const int mBase = blockIdx.x * 128;
  const int nBase = blockIdx.y * 64;

  const _Float16* A  = Aall + aStride * batch;
  const _Float16* Bh = (MODE == 0) ? (const _Float16*)nullptr : Bhall + bStride * batch;
  const float*    Bf = (MODE == 0) ? Bfall + bStride * batch : (const float*)nullptr;

  // double-buffered tiles: A [m][k], B [n][k] (N-major), rows padded to 40 halves
  __shared__ __align__(16) _Float16 sA[2][128][40];  // 20,480 B
  __shared__ __align__(16) _Float16 sB[2][64][40];   // 10,240 B

  const int tid  = (int)threadIdx.x;
  const int lane = tid & 31;
  const int wv   = tid >> 5;
  const int wm   = wv >> 1;        // 0..3  (M waves)
  const int wn   = wv & 1;         // 0..1  (N waves)
  const int lr   = lane & 15;
  const int hh   = lane >> 4;      // half-wave select

  // ---- tile stager (issues async copies and/or ds stores; completion via
  //      s_wait_asynccnt / s_wait_dscnt before the workgroup barrier) ----
  auto stage = [&](int b, int kk) {
    // A tile: 128 rows x 32 halves, contiguous 16B chunks -> async path
#pragma unroll
    for (int it = 0; it < 2; ++it) {
      int i = tid + it * 256;              // 512 uint4 granules
      int r = i >> 2, q = i & 3;
      async_copy16(A + (size_t)(mBase + r) * lda + kk + q * 8, &sA[b][r][q * 8]);
    }
    if (MODE == 1) {
      // B already N-major (rhs[d][x]) -> async contiguous copy
      int r = tid >> 2, q = tid & 3;       // 256 granules exactly
      async_copy16(Bh + (size_t)(nBase + r) * ldb + kk + q * 8, &sB[b][r][q * 8]);
    } else if (MODE == 2) {
      // K-major f16 source: transpose at half granularity via ds stores
      int k = tid >> 3, q = tid & 7;
      union { uint4 u; _Float16 h[8]; } t;
      t.u = *(const uint4*)(Bh + (size_t)(kk + k) * ldb + nBase + q * 8);
#pragma unroll
      for (int j = 0; j < 8; ++j) sB[b][q * 8 + j][k] = t.h[j];
    } else {
      // K-major f32 source (x): convert + transpose via ds stores
      int k = tid >> 3, q = tid & 7;
      const float* src = Bf + (size_t)(kk + k) * ldb + nBase + q * 8;
      float4 f0 = *(const float4*)(src);
      float4 f1 = *(const float4*)(src + 4);
      sB[b][q * 8 + 0][k] = (_Float16)f0.x; sB[b][q * 8 + 1][k] = (_Float16)f0.y;
      sB[b][q * 8 + 2][k] = (_Float16)f0.z; sB[b][q * 8 + 3][k] = (_Float16)f0.w;
      sB[b][q * 8 + 4][k] = (_Float16)f1.x; sB[b][q * 8 + 5][k] = (_Float16)f1.y;
      sB[b][q * 8 + 6][k] = (_Float16)f1.z; sB[b][q * 8 + 7][k] = (_Float16)f1.w;
    }
  };

  v8f acc[2][2];
#pragma unroll
  for (int i = 0; i < 2; ++i)
#pragma unroll
    for (int j = 0; j < 2; ++j)
#pragma unroll
      for (int e = 0; e < 8; ++e) acc[i][j][e] = 0.f;

  // prologue: fill buffer 0
  stage(0, 0);
  wait_async0();
  __syncthreads();

  int buf = 0;
  for (int kt = 0; kt < K; kt += 32) {
    // prefetch next tile into the other buffer (overlaps with WMMA below)
    if (kt + 32 < K) stage(buf ^ 1, kt + 32);

    // ---- fragment gather per the 16x16x32 f16 VGPR layouts ----
    v16h a[2], b[2];
#pragma unroll
    for (int mi = 0; mi < 2; ++mi) {
      int row = wm * 32 + mi * 16 + lr;
      // lanes 0-15: K 0-7 & 16-23 ; lanes 16-31: K 8-15 & 24-31
      v8h lo = *(const v8h*)&sA[buf][row][hh * 8];
      v8h hi = *(const v8h*)&sA[buf][row][hh * 8 + 16];
      a[mi] = mk16(lo, hi);
    }
#pragma unroll
    for (int ni = 0; ni < 2; ++ni) {
      int row = wn * 32 + ni * 16 + lr;
      // lanes 0-15: K 0-7 & 16-23 ; lanes 16-31: K 8-15 & 24-31 ([n][k] LDS)
      v8h lo = *(const v8h*)&sB[buf][row][hh * 8];
      v8h hi = *(const v8h*)&sB[buf][row][hh * 8 + 16];
      b[ni] = mk16(lo, hi);
    }
#pragma unroll
    for (int mi = 0; mi < 2; ++mi)
#pragma unroll
      for (int ni = 0; ni < 2; ++ni)
        acc[mi][ni] = __builtin_amdgcn_wmma_f32_16x16x32_f16(
            false, a[mi], false, b[ni], (short)0, acc[mi][ni], false, false);

    // our async prefetch must have landed before the group-wide barrier;
    // ds-store staging is covered by the s_wait_dscnt of the barrier itself
    wait_async0();
    __syncthreads();
    buf ^= 1;
  }

  // ---- fused epilogues ----
#pragma unroll
  for (int mi = 0; mi < 2; ++mi) {
#pragma unroll
    for (int ni = 0; ni < 2; ++ni) {
#pragma unroll
      for (int r = 0; r < 8; ++r) {
        int grow = mBase + wm * 32 + mi * 16 + hh * 8 + r;   // C/D: M = r + 8*hh
        int gcol = nBase + wn * 32 + ni * 16 + lr;           // C/D: N = lane&15
        float v = acc[mi][ni][r];
        if (MODE == 0) {
          float t3v = v + bp3[grow];
          float xv  = xf[((size_t)batch * C_DIM + grow) * HW_DIM + gcol];
          float t4v = __sinf(xv * PI_HALF);
          outH[((size_t)batch * C_DIM + grow) * HW_DIM + gcol] = (_Float16)fmaxf(t3v, t4v);
        } else if (MODE == 1) {
          float t = v * RS_HW * p7w[grow * C_DIM + gcol];
          // store transposed: t7t[d][c]
          outH[(size_t)batch * C_DIM * C_DIM + (size_t)gcol * C_DIM + grow] = (_Float16)t;
        } else {
          outF[((size_t)batch * C_DIM + grow) * HW_DIM + gcol] = v * RS_C;
        }
      }
    }
  }
}

extern "C" void kernel_launch(void* const* d_in, const int* in_sizes, int n_in,
                              void* d_out, int out_size, void* d_ws, size_t ws_size,
                              hipStream_t stream)
{
  const float* x   = (const float*)d_in[0];
  const float* Wp3 = (const float*)d_in[1];
  const float* bp3 = (const float*)d_in[2];
  const float* Wg1 = (const float*)d_in[3];
  const float* bg1 = (const float*)d_in[4];
  const float* Wg2 = (const float*)d_in[5];
  const float* bg2 = (const float*)d_in[6];
  const float* p7w = (const float*)d_in[7];
  float* out = (float*)d_out;

  char* ws = (char*)d_ws;
  const size_t halfCHW = (size_t)NBATCH * C_DIM * HW_DIM * sizeof(_Float16); // 51,380,224 B
  _Float16* lhs  = (_Float16*)(ws);
  _Float16* rhs  = (_Float16*)(ws + halfCHW);
  _Float16* t6h  = (_Float16*)(ws + 2 * halfCHW);
  _Float16* t7t  = (_Float16*)(ws + 3 * halfCHW);
  _Float16* wp3h = (_Float16*)(ws + 3 * halfCHW + (size_t)NBATCH * C_DIM * C_DIM * sizeof(_Float16));

  // 1) Wp3 -> f16
  cvt_f32_f16<<<dim3(256), dim3(256), 0, stream>>>(Wp3, wp3h, C_DIM * C_DIM);
  // 2) grouped convs -> lhs/rhs (f16)
  prep_kernel<<<dim3(49, C_DIM, NBATCH), dim3(64), 0, stream>>>(x, Wg1, bg1, Wg2, bg2, lhs, rhs);
  // 3) GEMM2: t7t = (p7w * (lhs @ rhs^T)/sqrt(hw))^T      M=256 N=256 K=3136
  gemm_wmma<1><<<dim3(2, 4, NBATCH), dim3(256), 0, stream>>>(
      lhs, rhs, nullptr, nullptr, nullptr, p7w, t7t, nullptr);
  // 4) GEMM1 + fused max/sin epilogue -> t6h              M=256 N=3136 K=256
  gemm_wmma<0><<<dim3(2, 49, NBATCH), dim3(256), 0, stream>>>(
      wp3h, nullptr, x, x, bp3, nullptr, t6h, nullptr);
  // 5) GEMM3 -> out (f32, already [n,C,H,W])              M=256 N=3136 K=256
  gemm_wmma<2><<<dim3(2, 49, NBATCH), dim3(256), 0, stream>>>(
      t7t, t6h, nullptr, nullptr, nullptr, nullptr, nullptr, out);
}
